// CausalPD_backbone_39410619908727
// MI455X (gfx1250) — matrix operand order repaired and checked by reading, not compiled, not run on hardware
//
#include <hip/hip_runtime.h>

// ---------------------------------------------------------------------------
// CausalPD backbone for MI455X (gfx1250, wave32, WMMA 16x16x32 bf16,
// async global->LDS staging for GEMM A-tiles)
// ---------------------------------------------------------------------------

typedef __bf16 bf16_t;
typedef __attribute__((ext_vector_type(8)))  __bf16 v8bf;
typedef __attribute__((ext_vector_type(16))) __bf16 v16bf;
typedef __attribute__((ext_vector_type(8)))  float  v8f;

#define B_   8
#define L_   1024
#define D_   1024
#define H_   16
#define DK_  64
#define LN_EPS 1e-5f
#define L2_EPS 1e-12f

// ---- CDNA5 async global->LDS (ASYNCcnt) -----------------------------------
static __device__ __forceinline__ void async_ld_b128(unsigned int lds_addr,
                                                     unsigned long long gaddr) {
  asm volatile("global_load_async_to_lds_b128 %0, %1, off"
               :: "v"(lds_addr), "v"(gaddr) : "memory");
}
static __device__ __forceinline__ void wait_async0() {
  asm volatile("s_wait_asynccnt 0x0" ::: "memory");
}

// ---- WMMA fragment loaders (CDNA5 ISA §7.12.2 layouts) ---------------
// A (16x32 bf16, MxK): lane<16 -> row=lane, K=kb..kb+7 (v0-3), kb+16..kb+23 (v4-7)
static __device__ __forceinline__ v16bf load_fragA(const bf16_t* base, int ld, int kb) {
  int lane = threadIdx.x & 31;
  const bf16_t* p = base + (lane & 15) * ld + kb + ((lane >> 4) << 3);
  v8bf lo = *(const v8bf*)p;
  v8bf hi = *(const v8bf*)(p + 16);
  v16bf r;
#pragma unroll
  for (int i = 0; i < 8; ++i) { r[i] = lo[i]; r[i + 8] = hi[i]; }
  return r;
}

// B (32x16 bf16, KxN), column n contiguous along K with stride ld
static __device__ __forceinline__ v16bf load_fragB(const bf16_t* base, int ld, int kb) {
  int lane = threadIdx.x & 31;
  const bf16_t* p = base + (lane & 15) * ld + kb + ((lane >> 4) << 3);
  v8bf lo = *(const v8bf*)p;
  v8bf hi = *(const v8bf*)(p + 16);
  v16bf r;
#pragma unroll
  for (int i = 0; i < 8; ++i) { r[i] = lo[i]; r[i + 8] = hi[i]; }
  return r;
}

static __device__ __forceinline__ v8f wmma_bf16(v16bf a, v16bf b, v8f c) {
  return __builtin_amdgcn_wmma_f32_16x16x32_bf16(false, a, false, b, (short)0, c,
                                                 false, false);
}

// ---- conversion kernels ---------------------------------------------------
__global__ void k_f32_to_bf16(const float* __restrict__ x, bf16_t* __restrict__ y, int n) {
  int i = blockIdx.x * blockDim.x + threadIdx.x;
  if (i < n) y[i] = (bf16_t)x[i];
}

// wt[n*D + k] = w[k*D + n]  (weights stored [out,in] -> contiguous B-frags)
__global__ void k_transpose_bf16(const float* __restrict__ w, bf16_t* __restrict__ wt) {
  int i = blockIdx.x * blockDim.x + threadIdx.x;  // i = n*D + k
  int n = i >> 10, k = i & (D_ - 1);
  wt[i] = (bf16_t)w[k * D_ + n];
}

// ---- projection GEMM core: block = one 16-row M-tile, A staged via async --
// Each of 8 waves owns a 128-wide N strip (8 sub-tiles of 16).
template <int MODE>  // 0: bf16 [B,H,L,DK], 1: bf16 [B,H,DK,L] (V), 2: f32 [B,L,D]
static __device__ __forceinline__ void gemm_body(
    const bf16_t* __restrict__ X, const bf16_t* __restrict__ WT,
    const float* __restrict__ bias, void* __restrict__ outp) {
  __shared__ bf16_t sA[16 * D_];   // 32 KB A tile
  int tid = threadIdx.x;
  int wave = tid >> 5, lane = tid & 31;
  int m0 = blockIdx.x << 4;

  // stage A tile (16 rows are contiguous in X) via async global->LDS
  unsigned int lbase = (unsigned int)(size_t)(&sA[0]);
  unsigned long long gbase = (unsigned long long)(size_t)(X + (size_t)m0 * D_);
#pragma unroll
  for (int i = 0; i < 8; ++i) {
    unsigned int off = (unsigned int)(tid + i * 256) * 16u;  // 16B chunks
    async_ld_b128(lbase + off, gbase + off);
  }
  wait_async0();
  __syncthreads();

  int nw0 = wave << 7;               // wave's N strip base
  v8f acc[8] = {};
#pragma unroll 2
  for (int kb = 0; kb < D_; kb += 32) {
    v16bf a = load_fragA(sA, D_, kb);            // from LDS
    __builtin_prefetch(WT + (size_t)nw0 * D_ + kb + 128, 0, 3);
    v16bf b[8];
#pragma unroll
    for (int j = 0; j < 8; ++j)                  // batch loads -> partial waits
      b[j] = load_fragB(WT + (size_t)(nw0 + j * 16) * D_, D_, kb);
#pragma unroll
    for (int j = 0; j < 8; ++j) acc[j] = wmma_bf16(a, b[j], acc[j]);
  }

  int rowhalf = (lane >> 4) << 3;
  int colsub  = lane & 15;
#pragma unroll
  for (int j = 0; j < 8; ++j) {
    int n  = nw0 + j * 16 + colsub;
    float bn = bias[n];
#pragma unroll
    for (int r = 0; r < 8; ++r) {
      int m = m0 + rowhalf + r;
      float v = acc[j][r] + bn;
      if (MODE == 2) {
        ((float*)outp)[(size_t)m * D_ + n] = v;
      } else {
        int bb = m >> 10, l = m & (L_ - 1);
        int h = n >> 6, dk = n & 63;
        if (MODE == 1)   // V: [B,H,DK,L] so PV B-frags are contiguous
          ((bf16_t*)outp)[((size_t)(bb * H_ + h) * DK_ + dk) * L_ + l] = (bf16_t)v;
        else             // Q,K: [B,H,L,DK]
          ((bf16_t*)outp)[((size_t)(bb * H_ + h) * L_ + l) * DK_ + dk] = (bf16_t)v;
      }
    }
  }
}

__global__ __launch_bounds__(256) void k_gemm_heads(
    const bf16_t* __restrict__ X, const bf16_t* __restrict__ WT,
    const float* __restrict__ bias, bf16_t* __restrict__ out) {
  gemm_body<0>(X, WT, bias, out);
}
__global__ __launch_bounds__(256) void k_gemm_headsT(
    const bf16_t* __restrict__ X, const bf16_t* __restrict__ WT,
    const float* __restrict__ bias, bf16_t* __restrict__ out) {
  gemm_body<1>(X, WT, bias, out);
}
__global__ __launch_bounds__(256) void k_gemm_f32(
    const bf16_t* __restrict__ X, const bf16_t* __restrict__ WT,
    const float* __restrict__ bias, float* __restrict__ out) {
  gemm_body<2>(X, WT, bias, out);
}

// ---- LayerNorm(ctx head) + row * bilinear ---------------------------------
__global__ __launch_bounds__(256) void k_ln_bil(
    const float* __restrict__ ctx, const float* __restrict__ bil,
    const float* __restrict__ gamma, const float* __restrict__ beta,
    bf16_t* __restrict__ c_out, bf16_t* __restrict__ cb_out) {
  __shared__ float crow[8][64];
  int wave = threadIdx.x >> 5, lane = threadIdx.x & 31;
  int rid = blockIdx.x * 8 + wave;     // (b*L + l)*H + h
  int h = rid & (H_ - 1);
  int bl = rid >> 4;
  const float* x = ctx + (size_t)bl * D_ + h * DK_;
  float x0 = x[lane], x1 = x[lane + 32];
  float s = x0 + x1;
#pragma unroll
  for (int m = 16; m >= 1; m >>= 1) s += __shfl_xor(s, m, 32);
  float mu = s * (1.f / 64.f);
  float d0 = x0 - mu, d1 = x1 - mu;
  float vs = d0 * d0 + d1 * d1;
#pragma unroll
  for (int m = 16; m >= 1; m >>= 1) vs += __shfl_xor(vs, m, 32);
  float inv = rsqrtf(vs * (1.f / 64.f) + LN_EPS);
  float c0 = d0 * inv * gamma[lane] + beta[lane];
  float c1 = d1 * inv * gamma[lane + 32] + beta[lane + 32];
  int b = bl >> 10, l = bl & (L_ - 1);
  size_t obase = ((size_t)(b * H_ + h) * L_ + l) * DK_;
  c_out[obase + lane]      = (bf16_t)c0;
  c_out[obase + lane + 32] = (bf16_t)c1;
  crow[wave][lane] = c0;
  crow[wave][lane + 32] = c1;
  __syncthreads();
  float a0 = 0.f, a1 = 0.f;
#pragma unroll 8
  for (int e = 0; e < 64; ++e) {
    float ce = crow[wave][e];
    a0 += ce * bil[e * 64 + lane];
    a1 += ce * bil[e * 64 + lane + 32];
  }
  cb_out[obase + lane]      = (bf16_t)a0;
  cb_out[obase + lane + 32] = (bf16_t)a1;
}

// ---- fused attention: scores+bias WMMA -> softmax/L2 fuse -> PV WMMA ------
__global__ __launch_bounds__(256) void k_attn(
    const bf16_t* __restrict__ qh, const bf16_t* __restrict__ kh,
    const bf16_t* __restrict__ vT, const bf16_t* __restrict__ ch,
    const bf16_t* __restrict__ cbil, const float* __restrict__ scale_p,
    const float* __restrict__ bscale_p, bf16_t* __restrict__ o1) {
  extern __shared__ char smemraw[];
  float*  sS = (float*)smemraw;             // 16x1024 f32 scores   (64 KB)
  float*  sB = (float*)(smemraw + 65536);   // 16x1024 f32 bias     (64 KB)
  bf16_t* sA = (bf16_t*)sB;                 // reuses sB once consumed (32 KB)

  int wave = threadIdx.x >> 5, lane = threadIdx.x & 31;
  int q0 = blockIdx.x << 4;
  int bh = blockIdx.y;                      // b*H + h
  float scale  = *scale_p;
  float bscale = *bscale_p;
  const bf16_t* qbase  = qh   + ((size_t)bh * L_ + q0) * DK_;
  const bf16_t* cbbase = cbil + ((size_t)bh * L_ + q0) * DK_;
  int rowhalf = (lane >> 4) << 3, colsub = lane & 15;

  // A fragments are invariant across column tiles -> hoist (4 loads total)
  v16bf aq0 = load_fragA(qbase,  DK_, 0), aq1 = load_fragA(qbase,  DK_, 32);
  v16bf ac0 = load_fragA(cbbase, DK_, 0), ac1 = load_fragA(cbbase, DK_, 32);

  // phase 1: each wave owns a 128-wide column strip (fully unrolled so the
  // scheduler can hoist next tile's B-fragment loads over current WMMAs)
#pragma unroll
  for (int ct = 0; ct < 8; ++ct) {
    int col0 = wave * 128 + ct * 16;
    const bf16_t* kb_ = kh + ((size_t)bh * L_ + col0) * DK_;
    const bf16_t* cb_ = ch + ((size_t)bh * L_ + col0) * DK_;
    v16bf bk0 = load_fragB(kb_, DK_, 0);
    v16bf bk1 = load_fragB(kb_, DK_, 32);
    v16bf bc0 = load_fragB(cb_, DK_, 0);
    v16bf bc1 = load_fragB(cb_, DK_, 32);
    v8f accS = {}, accB = {};
    accS = wmma_bf16(aq0, bk0, accS);
    accS = wmma_bf16(aq1, bk1, accS);
    accB = wmma_bf16(ac0, bc0, accB);
    accB = wmma_bf16(ac1, bc1, accB);
    int col = col0 + colsub;
#pragma unroll
    for (int r = 0; r < 8; ++r) {
      int row = rowhalf + r;
      sS[row * 1024 + col] = (col > q0 + row) ? -__builtin_inff() : accS[r] * scale;
      sB[row * 1024 + col] = accB[r];   // bias is NOT causally masked (ref)
    }
  }
  __syncthreads();

  // phase 2: row-wise softmax + L2(bias) + combine + L2(attn); 16 lanes/row
  int row = threadIdx.x >> 4, sub = threadIdx.x & 15;
  float* rs = sS + row * 1024;
  float* rb = sB + row * 1024;
  float mx = -__builtin_inff(), bsq = 0.f;
  for (int i = sub; i < 1024; i += 16) {
    mx = fmaxf(mx, rs[i]);
    float bv = rb[i];
    bsq += bv * bv;
  }
#pragma unroll
  for (int m = 8; m >= 1; m >>= 1) {
    mx = fmaxf(mx, __shfl_xor(mx, m, 32));
    bsq += __shfl_xor(bsq, m, 32);
  }
  float se = 0.f;
  for (int i = sub; i < 1024; i += 16) se += __expf(rs[i] - mx);
#pragma unroll
  for (int m = 8; m >= 1; m >>= 1) se += __shfl_xor(se, m, 32);
  float inv_se = 1.f / se;
  float inv_bn = bscale / fmaxf(sqrtf(bsq), L2_EPS);
  float asq = 0.f;
  for (int i = sub; i < 1024; i += 16) {
    float a = __expf(rs[i] - mx) * inv_se + rb[i] * inv_bn;
    rs[i] = a;
    asq += a * a;
  }
#pragma unroll
  for (int m = 8; m >= 1; m >>= 1) asq += __shfl_xor(asq, m, 32);
  float inv_an = 1.f / fmaxf(sqrtf(asq), L2_EPS);
  __syncthreads();                       // all sB reads done before aliasing
  for (int i = sub; i < 1024; i += 16) sA[row * 1024 + i] = (bf16_t)(rs[i] * inv_an);
  __syncthreads();

  // phase 3: out[16x64] = attn[16x1024] @ V[1024x64]; waves 0..3 own N-tiles
  if (wave < 4) {
    int n0 = wave << 4;
    const bf16_t* vb = vT + ((size_t)bh * DK_ + n0) * L_;
    v8f acc = {};
#pragma unroll 4
    for (int kb2 = 0; kb2 < L_; kb2 += 32) {
      v16bf a = load_fragA(sA, 1024, kb2);         // LDS (ds_load)
      v16bf b = load_fragB(vb, L_, kb2);           // contiguous (V transposed)
      acc = wmma_bf16(a, b, acc);
    }
    int b_ = bh >> 4, h = bh & (H_ - 1);
#pragma unroll
    for (int r = 0; r < 8; ++r) {
      int l = q0 + rowhalf + r;
      o1[((size_t)(b_ * L_ + l)) * D_ + h * DK_ + n0 + colsub] = (bf16_t)acc[r];
    }
  }
}

// ---------------------------------------------------------------------------
extern "C" void kernel_launch(void* const* d_in, const int* in_sizes, int n_in,
                              void* d_out, int out_size, void* d_ws, size_t ws_size,
                              hipStream_t stream) {
  (void)in_sizes; (void)n_in; (void)out_size; (void)ws_size;
  const float* Q      = (const float*)d_in[0];
  const float* ctx    = (const float*)d_in[1];
  // d_in[2] = attn_mask (causal; applied analytically)
  const float* W_q    = (const float*)d_in[3];
  const float* b_q    = (const float*)d_in[4];
  const float* W_k    = (const float*)d_in[5];
  const float* b_k    = (const float*)d_in[6];
  const float* W_v    = (const float*)d_in[7];
  const float* b_v    = (const float*)d_in[8];
  const float* bil    = (const float*)d_in[9];
  const float* gamma  = (const float*)d_in[10];
  const float* beta   = (const float*)d_in[11];
  const float* scale  = (const float*)d_in[12];
  const float* bscale = (const float*)d_in[13];
  const float* W_o    = (const float*)d_in[14];
  const float* b_o    = (const float*)d_in[15];
  float* out = (float*)d_out;

  char* ws = (char*)d_ws;
  size_t off = 0;
  auto alloc = [&](size_t bytes) {
    char* p = ws + off;
    off += (bytes + 255) & ~size_t(255);
    return p;
  };
  const size_t NBLD = (size_t)B_ * L_ * D_;          // 8M elems
  bf16_t* Qb   = (bf16_t*)alloc(NBLD * 2);
  bf16_t* WqT  = (bf16_t*)alloc((size_t)D_ * D_ * 2);
  bf16_t* WkT  = (bf16_t*)alloc((size_t)D_ * D_ * 2);
  bf16_t* WvT  = (bf16_t*)alloc((size_t)D_ * D_ * 2);
  bf16_t* WoT  = (bf16_t*)alloc((size_t)D_ * D_ * 2);
  bf16_t* qhd  = (bf16_t*)alloc(NBLD * 2);           // [B,H,L,DK]
  bf16_t* khd  = (bf16_t*)alloc(NBLD * 2);           // [B,H,L,DK]
  bf16_t* vTd  = (bf16_t*)alloc(NBLD * 2);           // [B,H,DK,L]
  bf16_t* chd  = (bf16_t*)alloc(NBLD * 2);           // [B,H,L,DK]
  bf16_t* cbd  = (bf16_t*)alloc(NBLD * 2);           // [B,H,L,DK]
  bf16_t* o1   = (bf16_t*)alloc(NBLD * 2);           // [B,L,D]

  // 1) precision conversion + weight transposition
  k_f32_to_bf16<<<(int)(NBLD / 256), 256, 0, stream>>>(Q, Qb, (int)NBLD);
  k_transpose_bf16<<<D_ * D_ / 256, 256, 0, stream>>>(W_q, WqT);
  k_transpose_bf16<<<D_ * D_ / 256, 256, 0, stream>>>(W_k, WkT);
  k_transpose_bf16<<<D_ * D_ / 256, 256, 0, stream>>>(W_v, WvT);
  k_transpose_bf16<<<D_ * D_ / 256, 256, 0, stream>>>(W_o, WoT);

  // 2) QKV projections (one block per 16-row M-tile; A tile async-staged)
  k_gemm_heads <<<512, 256, 0, stream>>>(Qb, WqT, b_q, qhd);
  k_gemm_heads <<<512, 256, 0, stream>>>(Qb, WkT, b_k, khd);
  k_gemm_headsT<<<512, 256, 0, stream>>>(Qb, WvT, b_v, vTd);

  // 3) LayerNorm(ctx) + bilinear row product
  k_ln_bil<<<(B_ * L_ * H_) / 8, 256, 0, stream>>>(ctx, bil, gamma, beta, chd, cbd);

  // 4) fused attention
  dim3 agrid(L_ / 16, B_ * H_);
  k_attn<<<agrid, 256, 131072, stream>>>(qhd, khd, vTd, chd, cbd, scale, bscale, o1);

  // 5) output projection
  k_gemm_f32<<<512, 256, 0, stream>>>(o1, WoT, b_o, out);
}
